// CapsuleLayer_45724221833633
// MI455X (gfx1250) — matrix operand order repaired and checked by reading, not compiled, not run on hardware
//
#include <hip/hip_runtime.h>
#include <hip/hip_bf16.h>

// ---------------------------------------------------------------------------
// CapsuleLayer on MI455X (gfx1250), collapsed to:
//   one 5x5 conv (Cin=64, Cout=256) via implicit GEMM on bf16 WMMA
//   + position-dependent scale r(y,x) + squash over groups of 32 channels.
// Compute-bound: 53.7 GFLOP vs ~94MB HBM traffic (~4us @ 23.3 TB/s).
// Input tiles come in via the Tensor Data Mover (one tensor_load_to_lds per
// workgroup from a zero-padded channels-last bf16 tensor); B-fragment loads
// are grouped ahead of the 4 WMMAs per k-step.
// ---------------------------------------------------------------------------

typedef __attribute__((ext_vector_type(16))) __bf16 v16bf;
typedef __attribute__((ext_vector_type(8)))  __bf16 v8bf;
typedef __attribute__((ext_vector_type(8)))  float  v8f;
typedef __attribute__((ext_vector_type(4)))  unsigned int u32x4;
typedef __attribute__((ext_vector_type(8)))  int i32x8;
typedef __attribute__((ext_vector_type(4)))  int i32x4;
typedef __attribute__((address_space(3)))    char lds_char;

#define HH   128
#define WW   128
#define CIN  64
#define COUT 256
#define KS   5
#define MTILE 32
#define COLS (MTILE + 4)      // 36 staged columns (halo of 2 each side)
#define PW   (WW + 4)         // 132 padded width
#define PH   (HH + 4)         // 132 padded height
#define RES_STRIDE 258        // padded f32 row stride (kills LDS bank conflicts)

__device__ __forceinline__ unsigned short f32_to_bf16(float f) {
    unsigned int x = __float_as_uint(f);
    unsigned int r = (x + 0x7FFFu + ((x >> 16) & 1u)) >> 16;   // RNE
    return (unsigned short)r;
}

// ---------------------------------------------------------------------------
// Pack W [4][256][16][5][5] f32 -> bf16 B-fragments, k contiguous per lane half:
//   flat index = ((((kk*2 + cinb)*16 + coutb)*32 + lane)*16 + e)
// B(32x16), this code's k order (same as A): lanes 0-15 = N, K=0..15 (2/VGPR); lanes 16-31 K=16..31.
// ---------------------------------------------------------------------------
__global__ void caps_pack_weights(const float* __restrict__ W,
                                  unsigned short* __restrict__ wf) {
    int idx   = blockIdx.x * 256 + threadIdx.x;   // < 25*2*16*32*16 = 409600
    int e     = idx & 15;
    int lane  = (idx >> 4) & 31;
    int coutb = (idx >> 9) & 15;
    int cinb  = (idx >> 13) & 1;
    int kk    = idx >> 14;                        // 0..24
    int cout  = coutb * 16 + (lane & 15);
    int cin   = cinb * 32 + (lane >> 4) * 16 + e; // K index within 64
    int ic = cin >> 4, ii = cin & 15;
    float w = W[(((ic * 256 + cout) * 16 + ii) * 25) + kk];
    wf[idx] = f32_to_bf16(w);
}

__global__ void caps_bias_sum(const float* __restrict__ bias,
                              float* __restrict__ bs) {
    int c = threadIdx.x;                          // 256 threads
    bs[c] = bias[c] + bias[256 + c] + bias[512 + c] + bias[768 + c];
}

// ---------------------------------------------------------------------------
// Transpose + pad + convert: u [B][64][128][128] f32  ->
//   upad [B][132][132][64] bf16, zero halo of 2 on every side.
// One block per (b, padded row). Reads coalesced along x, LDS flip,
// writes fully coalesced (ch fastest = contiguous bf16).
// ---------------------------------------------------------------------------
__global__ __launch_bounds__(256)
void caps_pad_transpose(const float* __restrict__ u,
                        unsigned short* __restrict__ upad) {
    __shared__ unsigned short t16[WW * CIN];      // [x][ch] bf16, 16KB
    int bid = blockIdx.x;                         // 4*132 = 528
    int yp  = bid % PH;
    int b   = bid / PH;
    int y   = yp - 2;
    int tid = threadIdx.x;
    bool interior = (unsigned)y < (unsigned)HH;
    if (interior) {
        const float* ub = u + ((size_t)b * CIN) * (HH * WW) + (size_t)y * WW;
        for (int e = tid; e < CIN * WW; e += 256) {
            int ch = e >> 7, x = e & 127;
            t16[x * CIN + ch] = f32_to_bf16(ub[(size_t)ch * (HH * WW) + x]);
        }
    } else {
        for (int e = tid; e < CIN * WW; e += 256) t16[e] = 0;
    }
    __syncthreads();
    unsigned short* orow = upad + ((size_t)b * PH + yp) * (PW * CIN);
    for (int e = tid; e < PW * CIN; e += 256) {
        int xp = e >> 6, ch = e & 63;
        int x = xp - 2;
        unsigned short v = 0;
        if (interior && (unsigned)x < (unsigned)WW) v = t16[x * CIN + ch];
        orow[e] = v;
    }
}

// ---------------------------------------------------------------------------
// Main: one workgroup = 32 consecutive x at fixed (b,y), all 256 couts.
// Input tile [5][36][64] bf16 arrives via one TDM tensor_load_to_lds.
// 8 waves = 2 m-groups x 4 n-quarters; 50 K-steps x 4 WMMAs per wave.
// ---------------------------------------------------------------------------
__global__ __launch_bounds__(256)
void caps_main(const unsigned short* __restrict__ upad,
               const unsigned short* __restrict__ wfrag,
               const float* __restrict__ bias_sum,
               float* __restrict__ out) {
    // 33024B LDS: bf16 input tile [5][36][64] (23040B) reused as f32 result.
    __shared__ float4 smem4[(MTILE * RES_STRIDE * 4) / 16];
    unsigned short* in_tile = (unsigned short*)smem4;
    float*          res     = (float*)smem4;

    int tid = threadIdx.x;
    int bid = blockIdx.x;                 // B*H*(W/MTILE) = 4*128*4 = 2048
    int xt  = bid & 3;
    int y   = (bid >> 2) & 127;
    int b   = bid >> 9;
    int x0  = xt * MTILE;

    // ---- TDM: DMA the [5 rows][36 cols][64 ch] bf16 tile into LDS --------
    // 2D tensor view of upad row-space: dim0 = PW*CIN elems, data_size = 8B.
    // tile row = 36*64*2B = 4608B = 576 units; line stride = 132*64*2B/8 = 2112.
    // Tile origin (padded coords): row = y, col = x0 -> always fully in-bounds.
    if (tid < 32) {                        // wave 0 issues; EXEC-independent op
        unsigned lds_off = (unsigned)(unsigned long long)(lds_char*)in_tile;
        unsigned long long gaddr = (unsigned long long)(const void*)upad
            + ((unsigned long long)((b * PH + y) * PW + x0) * CIN) * 2ull;
        u32x4 g0v = { 1u,                               // count=1 valid D#
                      lds_off,                          // [63:32] lds_addr
                      (unsigned)gaddr,                  // [95:64] gaddr lo
                      (unsigned)((gaddr >> 32) & 0x01FFFFFFull) | 0x80000000u };
                                                        // [120:96] + type=2
        i32x8 g1v = { 0x30000,       // data_size=3 (8B), mask=0, no flags
                      0x08400000,    // tensor_dim0 = 2112  (bits 63:48)
                      0x00840000,    // tensor_dim1 = 132   (bits 111:80)
                      0x02400000,    // tile_dim0   = 576   (bits 127:112)
                      5,             // tile_dim1   = 5, tile_dim2 = 0
                      2112,          // tensor_dim0_stride lo
                      0, 0 };        // stride hi, dim1_stride
        i32x4 z4 = { 0, 0, 0, 0 };
        i32x8 z8 = { 0, 0, 0, 0, 0, 0, 0, 0 };
        __builtin_amdgcn_tensor_load_to_lds(g0v, g1v, z4, z4, z8, 0);
        __builtin_amdgcn_s_wait_tensorcnt(0);
    }
    __syncthreads();

    int lane   = tid & 31;
    int wid    = tid >> 5;
    int mg     = wid >> 2;        // 0..1 : which 16-pixel group
    int nq     = wid & 3;         // 0..3 : which 64-cout quarter
    int m      = lane & 15;
    int laneHi = lane >> 4;

    v8f acc[4];
#pragma unroll
    for (int t = 0; t < 4; ++t) acc[t] = (v8f){0.f,0.f,0.f,0.f,0.f,0.f,0.f,0.f};

    const v16bf* wf = (const v16bf*)wfrag;

    for (int kh = 0; kh < KS; ++kh) {
        for (int kw = 0; kw < KS; ++kw) {
            int kk  = kh * KS + kw;
            int col = mg * 16 + m + kw;
#pragma unroll
            for (int cinb = 0; cinb < 2; ++cinb) {
                // Group all 4 B fragments first: one clause of 8
                // global_load_b128, one wait, then 4 back-to-back WMMAs.
                const v16bf* wfk =
                    wf + (size_t)(((kk * 2 + cinb) * 16 + nq * 4) * 32 + lane);
                v16bf bfr[4];
#pragma unroll
                for (int t = 0; t < 4; ++t) bfr[t] = wfk[t * 32];
                // A (16x32 bf16): lanes 0-15 K 0..15, lanes 16-31
                // K 16..31 -> two 16B ds_load_b128 at +0 and +16B.
                int abase = (kh * COLS + col) * CIN + cinb * 32 + laneHi * 16;
                union { v8bf h[2]; v16bf v; } a;
                a.h[0] = *(const v8bf*)(const void*)(in_tile + abase);
                a.h[1] = *(const v8bf*)(const void*)(in_tile + abase + 8);
#pragma unroll
                for (int t = 0; t < 4; ++t) {
                    acc[t] = __builtin_amdgcn_wmma_f32_16x16x32_bf16(
                        false, a.v, false, bfr[t], (short)0, acc[t], false, false);
                }
            }
        }
    }
    __syncthreads();   // done reading in_tile; reuse LDS for results

    // C/D layout: N = lane&15, M = vgpr + 8*(lane>>4)
#pragma unroll
    for (int t = 0; t < 4; ++t) {
        int cout = (nq * 4 + t) * 16 + m;
#pragma unroll
        for (int i = 0; i < 8; ++i) {
            int xl = mg * 16 + laneHi * 8 + i;
            res[xl * RES_STRIDE + cout] = acc[t][i];
        }
    }
    __syncthreads();

    // ---- epilogue: r(y,x) scale + squash over od=32, coalesced stores ----
    int xl = tid & 31;            // lanes span x -> 128B coalesced stores
    int oc = tid >> 5;            // one capsule group per wave
    int x  = x0 + xl;
    int rows = min(y + 2, HH - 1) - max(y - 2, 0) + 1;
    int cols = min(x + 2, WW - 1) - max(x - 2, 0) + 1;
    float r = 1.f / (8.f * (float)rows * (float)cols);

    const float* rrow = res + xl * RES_STRIDE + oc * 32;
    const float* bsum = bias_sum + oc * 32;
    float sq = 0.f;
#pragma unroll
    for (int od = 0; od < 32; ++od) {
        float p = r * (rrow[od] + bsum[od]);
        sq += p * p;
    }
    float scale = sq / ((1.f + sq) * sqrtf(sq + 1e-9f));
    float* ob = out + (((size_t)b * COUT + oc * 32) * HH + y) * WW + x;
#pragma unroll
    for (int od = 0; od < 32; ++od) {
        float p = r * (rrow[od] + bsum[od]);
        ob[(size_t)od * (HH * WW)] = scale * p;
    }
}

// ---------------------------------------------------------------------------
extern "C" void kernel_launch(void* const* d_in, const int* in_sizes, int n_in,
                              void* d_out, int out_size, void* d_ws, size_t ws_size,
                              hipStream_t stream) {
    const float* u    = (const float*)d_in[0];   // [4,4,16,128,128]
    const float* W    = (const float*)d_in[1];   // [4,256,16,5,5]
    const float* bias = (const float*)d_in[2];   // [4,256]
    float* out = (float*)d_out;                  // [4,8,32,128,128]

    // workspace: wfrag 819200B | bias_sum 1024B | upad 8,921,088B (~9.75MB)
    unsigned short* wfrag = (unsigned short*)d_ws;
    float* bsum           = (float*)((char*)d_ws + 819200);
    unsigned short* upad  = (unsigned short*)((char*)d_ws + 820224);

    caps_pack_weights<<<1600, 256, 0, stream>>>(W, wfrag);
    caps_bias_sum<<<1, 256, 0, stream>>>(bias, bsum);
    caps_pad_transpose<<<4 * PH, 256, 0, stream>>>(u, upad);
    caps_main<<<2048, 256, 0, stream>>>(upad, wfrag, bsum, out);
}
